// MergedBDH_84894323573384
// MI455X (gfx1250) — matrix operand order repaired and checked by reading, not compiled, not run on hardware
//
#include <hip/hip_runtime.h>
#include <hip/hip_bf16.h>

#define T_  4096
#define D_  256
#define NH_ 4
#define N_  256
#define NL_ 4
#define V_  256
#define HN_ (NH_*N_)   // 1024
#define QB_ 8          // query blocks (of 16 rows) per workgroup in attention
#define TWO_PI_ 6.28318530717958647692f

typedef __attribute__((ext_vector_type(16))) __bf16 v16bf;
typedef __attribute__((ext_vector_type(8)))  __bf16 v8bf;
typedef __attribute__((ext_vector_type(8)))  float  v8f;

__device__ __forceinline__ v8f wmma_bf16(v16bf a, v16bf b, v8f c) {
  // D = A(16x32 bf16) * B(32x16 bf16) + C(16x16 f32)
  return __builtin_amdgcn_wmma_f32_16x16x32_bf16(
      /*neg_a=*/false, a, /*neg_b=*/false, b,
      /*c_mod=*/(short)0, c, /*reuse_a=*/false, /*reuse_b=*/false);
}

union frag_u { v16bf v; v8bf h[2]; };

// A fragment: 16x32 tile, row-major source, per ISA 16-bit A layout:
//   lane<16 : row=lane,    K = {0..7, 16..23}
//   lane>=16: row=lane-16, K = {8..15, 24..31}
__device__ __forceinline__ v16bf load_frag_a(const __bf16* src, int ld, int lane) {
  int half = lane >> 4, m = lane & 15;
  const __bf16* p = src + m * ld + 8 * half;
  frag_u u;
  u.h[0] = *(const v8bf*)(p);
  u.h[1] = *(const v8bf*)(p + 16);
  return u.v;
}

// B fragment for C = A @ Bt^T: Bt is row-major [N,K]; per ISA 16-bit B layout:
//   lane<16 : col=lane,    K = {0..7, 16..23} ; lane>=16: col=lane-16, K = {8..15, 24..31}
__device__ __forceinline__ v16bf load_frag_b(const __bf16* src, int ld, int lane) {
  int half = lane >> 4, n = lane & 15;
  const __bf16* p = src + n * ld + 8 * half;
  frag_u u;
  u.h[0] = *(const v8bf*)(p);
  u.h[1] = *(const v8bf*)(p + 16);
  return u.v;
}

// Async global -> LDS copy of 16 bytes per lane (CDNA5, tracked by ASYNCcnt).
// VDST carries the LDS byte address (low 32 bits of the generic LDS pointer,
// per the LDS aperture truncation rule); VADDR is the 64-bit global address.
__device__ __forceinline__ void async_g2l_b128(unsigned lds_off, const void* gaddr) {
  asm volatile("global_load_async_to_lds_b128 %0, %1, off"
               :: "v"(lds_off), "v"(gaddr) : "memory");
}
__device__ __forceinline__ void wait_asynccnt0() {
  asm volatile("s_wait_asynccnt 0x0" ::: "memory");
}

// ---------------------------------------------------------------------------
// Prep: transpose f32 weights into bf16 "Bt" layout: out[(h*N+n)*K + k] = in[(h*K+k)*N + n]
__global__ void prep_wT_kernel(const float* __restrict__ in, __bf16* __restrict__ out,
                               int H, int K, int N) {
  int i = blockIdx.x * blockDim.x + threadIdx.x;
  if (i >= H * K * N) return;
  int h = i / (K * N), r = i % (K * N), k = r / N, n = r % N;
  out[((size_t)h * N + n) * K + k] = (__bf16)in[i];
}

// ---------------------------------------------------------------------------
// Embed + LayerNorm -> x (f32), xb (bf16), xT (bf16 transposed [D,T])
__global__ void embed_ln_kernel(const int* __restrict__ idx, const float* __restrict__ embed,
                                float* __restrict__ x, __bf16* __restrict__ xb,
                                __bf16* __restrict__ xT) {
  int t = blockIdx.x, d = threadIdx.x;            // blockDim.x == 256 == D_
  float v = embed[(size_t)idx[t] * D_ + d];
  float s = v, q = v * v;
#pragma unroll
  for (int off = 1; off < 32; off <<= 1) {
    s += __shfl_xor(s, off, 32);
    q += __shfl_xor(q, off, 32);
  }
  __shared__ float ss[8], sq[8];
  int w = threadIdx.x >> 5;
  if ((threadIdx.x & 31) == 0) { ss[w] = s; sq[w] = q; }
  __syncthreads();
  s = 0.f; q = 0.f;
#pragma unroll
  for (int i = 0; i < 8; ++i) { s += ss[i]; q += sq[i]; }
  float mu = s * (1.f / D_);
  float var = q * (1.f / D_) - mu * mu;
  float nv = (v - mu) * rsqrtf(var + 1e-5f);
  x [(size_t)t * D_ + d] = nv;
  xb[(size_t)t * D_ + d] = (__bf16)nv;
  xT[(size_t)d * T_ + t] = (__bf16)nv;
}

// ---------------------------------------------------------------------------
// x_sparse = relu(x @ enc_h); QR = rope(x_sparse). One 16x16 C tile per wave.
__global__ void enc_rope_kernel(const __bf16* __restrict__ xb, const __bf16* __restrict__ encT,
                                const float* __restrict__ freqs,
                                __bf16* __restrict__ xs, __bf16* __restrict__ qr) {
  int lane = threadIdx.x & 31, w = threadIdx.x >> 5;
  int tile = blockIdx.x * 8 + w;                  // NH * (T/16) * (N/16) tiles
  int h  = tile >> 12;                            // 4096 tiles per head
  int r  = tile & 4095;
  int it = r >> 4, nt = r & 15;
  const __bf16* A  = xb + (size_t)(it * 16) * D_;
  const __bf16* Bt = encT + ((size_t)h * N_ + nt * 16) * D_;
  v8f c = {};
#pragma unroll
  for (int kt = 0; kt < 8; ++kt)
    c = wmma_bf16(load_frag_a(A + kt * 32, D_, lane),
                  load_frag_b(Bt + kt * 32, D_, lane), c);
  int nl = lane & 15, half = lane >> 4;
  int n = nt * 16 + nl;
  float f = freqs[n];
#pragma unroll
  for (int v = 0; v < 8; ++v) {
    int t = it * 16 + half * 8 + v;
    float sv = fmaxf(c[v], 0.f);                  // relu
    float other = __shfl_xor(sv, 1, 32);          // pair partner (n ^ 1)
    float vrot = (n & 1) ? other : -other;
    float p = (float)t * f;
    p = (p - floorf(p)) * TWO_PI_;
    float qv = sv * cosf(p) + vrot * sinf(p);
    size_t o = ((size_t)h * T_ + t) * N_ + n;
    xs[o] = (__bf16)sv;
    qr[o] = (__bf16)qv;
  }
}

// ---------------------------------------------------------------------------
// Fused attention: yKV = LN( tril(QR @ QR^T, -1) @ x ).
// One workgroup = 8 waves = 8 query blocks (128 rows) of one head.
// Each 32-key step: async-stage QR slab (32x256) and xT slab (256x32) into
// LDS once per workgroup, then every wave runs 8+8 QK WMMAs and 16 SV WMMAs
// out of LDS. S tiles round-trip per-wave LDS to become bf16 A fragments.
__global__ void attn_kernel(const __bf16* __restrict__ qr, const __bf16* __restrict__ xT,
                            __bf16* __restrict__ ykv) {
  __shared__ __bf16 sQR[32 * N_];                 // 32 keys x 256 dims (16KB)
  __shared__ __bf16 sXT[D_ * 32];                 // 256 dims x 32 keys (16KB)
  __shared__ __bf16 sS[8][16 * 32];               // per-wave S tile (8KB)
  int lane = threadIdx.x & 31, w = threadIdx.x >> 5;
  int h   = blockIdx.x >> 5;                      // T/(16*QB_) = 32 superblocks/head
  int sb  = blockIdx.x & 31;
  int qb0 = sb * QB_;
  int ib  = qb0 + w;                              // this wave's query block
  const __bf16* Qb = qr + (size_t)h * T_ * N_;
  int nl = lane & 15, half = lane >> 4;

  v16bf aq[8];                                    // cached query-row A fragments
#pragma unroll
  for (int kt = 0; kt < 8; ++kt)
    aq[kt] = load_frag_a(Qb + (size_t)(ib * 16) * N_ + kt * 32, N_, lane);

  v8f acc[16];
  v8f z = {};
#pragma unroll
  for (int dt = 0; dt < 16; ++dt) acc[dt] = z;

  unsigned sQR_off = (unsigned)(size_t)(&sQR[0]);
  unsigned sXT_off = (unsigned)(size_t)(&sXT[0]);
  __bf16* myS = &sS[w][0];
  int tid = threadIdx.x;
  int jbMax = qb0 + QB_ - 1;

  for (int jb = 0; jb <= jbMax; jb += 2) {
    // ---- cooperative async fill: 256 threads x 64B each per slab ----
    {
      int row = tid >> 3, ch = tid & 7;           // sQR: key row, 64B chunk
      const __bf16* g = Qb + (size_t)(jb * 16 + row) * N_ + ch * 32;
      unsigned l = sQR_off + (unsigned)(row * N_ + ch * 32) * 2u;
      async_g2l_b128(l,      g);
      async_g2l_b128(l + 16, g + 8);
      async_g2l_b128(l + 32, g + 16);
      async_g2l_b128(l + 48, g + 24);
      const __bf16* g2 = xT + (size_t)tid * T_ + jb * 16;  // sXT: dim row
      unsigned l2 = sXT_off + (unsigned)tid * 64u;
      async_g2l_b128(l2,      g2);
      async_g2l_b128(l2 + 16, g2 + 8);
      async_g2l_b128(l2 + 32, g2 + 16);
      async_g2l_b128(l2 + 48, g2 + 24);
    }
    wait_asynccnt0();
    __syncthreads();

    if (jb <= ib) {
      v8f s0 = {};
#pragma unroll
      for (int kt = 0; kt < 8; ++kt)
        s0 = wmma_bf16(aq[kt], load_frag_b(sQR + kt * 32, N_, lane), s0);
      v8f s1 = {};
      if (jb + 1 <= ib) {
#pragma unroll
        for (int kt = 0; kt < 8; ++kt)
          s1 = wmma_bf16(aq[kt], load_frag_b(sQR + 16 * N_ + kt * 32, N_, lane), s1);
      }
#pragma unroll
      for (int v = 0; v < 8; ++v) {               // tril(k=-1) mask on diagonal tile
        int m = half * 8 + v;
        if (jb == ib && m <= nl) s0[v] = 0.f;
        if (jb + 1 == ib && m <= nl) s1[v] = 0.f;
        myS[m * 32 + nl]      = (__bf16)s0[v];
        myS[m * 32 + 16 + nl] = (__bf16)s1[v];
      }
      v16bf as = load_frag_a(myS, 32, lane);      // S as A fragment (K = 32 keys)
#pragma unroll
      for (int dt = 0; dt < 16; ++dt)
        acc[dt] = wmma_bf16(as, load_frag_b(sXT + dt * 16 * 32, 32, lane), acc[dt]);
    }
    __syncthreads();                              // slabs reused next iteration
  }

  // In-register LayerNorm over D: per-row partials, then 16-lane shfl tree.
  float mu[8], rs[8];
#pragma unroll
  for (int v = 0; v < 8; ++v) {
    float s = 0.f, q = 0.f;
#pragma unroll
    for (int dt = 0; dt < 16; ++dt) { float f = acc[dt][v]; s += f; q += f * f; }
#pragma unroll
    for (int off = 1; off < 16; off <<= 1) {
      s += __shfl_xor(s, off, 32);
      q += __shfl_xor(q, off, 32);
    }
    float m = s * (1.f / D_);
    mu[v] = m;
    rs[v] = rsqrtf(q * (1.f / D_) - m * m + 1e-5f);
  }
#pragma unroll
  for (int v = 0; v < 8; ++v) {
    int t = ib * 16 + half * 8 + v;
#pragma unroll
    for (int dt = 0; dt < 16; ++dt) {
      int d = dt * 16 + nl;
      ykv[((size_t)h * T_ + t) * D_ + d] = (__bf16)((acc[dt][v] - mu[v]) * rs[v]);
    }
  }
}

// ---------------------------------------------------------------------------
// y_sparse = relu(yKV @ encV_h); xy[t][h*N+n] = x_sparse * y_sparse
__global__ void encv_gate_kernel(const __bf16* __restrict__ ykv, const __bf16* __restrict__ encVT,
                                 const __bf16* __restrict__ xs, __bf16* __restrict__ xy) {
  int lane = threadIdx.x & 31, w = threadIdx.x >> 5;
  int tile = blockIdx.x * 8 + w;
  int h  = tile >> 12;
  int r  = tile & 4095;
  int it = r >> 4, nt = r & 15;
  const __bf16* A  = ykv + ((size_t)h * T_ + it * 16) * D_;
  const __bf16* Bt = encVT + ((size_t)h * N_ + nt * 16) * D_;
  v8f c = {};
#pragma unroll
  for (int kt = 0; kt < 8; ++kt)
    c = wmma_bf16(load_frag_a(A + kt * 32, D_, lane),
                  load_frag_b(Bt + kt * 32, D_, lane), c);
  int nl = lane & 15, half = lane >> 4;
  int n = nt * 16 + nl;
#pragma unroll
  for (int v = 0; v < 8; ++v) {
    int t = it * 16 + half * 8 + v;
    float y = fmaxf(c[v], 0.f);
    float g = (float)xs[((size_t)h * T_ + t) * N_ + n];
    xy[(size_t)t * HN_ + h * N_ + n] = (__bf16)(g * y);
  }
}

// ---------------------------------------------------------------------------
// yMLP = xy @ decoder; x = LN(x + LN(yMLP)). One wave per 16-row block,
// accumulating the full 16x256 output so both LayerNorms stay in registers.
__global__ void mlp_ln_kernel(const __bf16* __restrict__ xy, const __bf16* __restrict__ decT,
                              float* __restrict__ x, __bf16* __restrict__ xb,
                              __bf16* __restrict__ xT) {
  int lane = threadIdx.x & 31, w = threadIdx.x >> 5;
  int it = blockIdx.x * 8 + w;                    // T/16 row blocks
  int nl = lane & 15, half = lane >> 4;
  v8f acc[16];
  v8f z = {};
#pragma unroll
  for (int dt = 0; dt < 16; ++dt) acc[dt] = z;
  const __bf16* A = xy + (size_t)(it * 16) * HN_;
  for (int kt = 0; kt < 32; ++kt) {               // K = 1024
    v16bf a = load_frag_a(A + kt * 32, HN_, lane);
#pragma unroll
    for (int dt = 0; dt < 16; ++dt)
      acc[dt] = wmma_bf16(a, load_frag_b(decT + (size_t)(dt * 16) * HN_ + kt * 32, HN_, lane),
                          acc[dt]);
  }
  float mu[8], rs[8];
#pragma unroll
  for (int v = 0; v < 8; ++v) {                   // LN(yMLP)
    float s = 0.f, q = 0.f;
#pragma unroll
    for (int dt = 0; dt < 16; ++dt) { float f = acc[dt][v]; s += f; q += f * f; }
#pragma unroll
    for (int off = 1; off < 16; off <<= 1) {
      s += __shfl_xor(s, off, 32);
      q += __shfl_xor(q, off, 32);
    }
    float m = s * (1.f / D_);
    mu[v] = m;
    rs[v] = rsqrtf(q * (1.f / D_) - m * m + 1e-5f);
  }
#pragma unroll
  for (int v = 0; v < 8; ++v) {                   // residual add
    int t = it * 16 + half * 8 + v;
#pragma unroll
    for (int dt = 0; dt < 16; ++dt) {
      int d = dt * 16 + nl;
      acc[dt][v] = x[(size_t)t * D_ + d] + (acc[dt][v] - mu[v]) * rs[v];
    }
  }
#pragma unroll
  for (int v = 0; v < 8; ++v) {                   // LN(x + LN(yMLP))
    float s = 0.f, q = 0.f;
#pragma unroll
    for (int dt = 0; dt < 16; ++dt) { float f = acc[dt][v]; s += f; q += f * f; }
#pragma unroll
    for (int off = 1; off < 16; off <<= 1) {
      s += __shfl_xor(s, off, 32);
      q += __shfl_xor(q, off, 32);
    }
    float m = s * (1.f / D_);
    mu[v] = m;
    rs[v] = rsqrtf(q * (1.f / D_) - m * m + 1e-5f);
  }
#pragma unroll
  for (int v = 0; v < 8; ++v) {
    int t = it * 16 + half * 8 + v;
#pragma unroll
    for (int dt = 0; dt < 16; ++dt) {
      int d = dt * 16 + nl;
      float nv = (acc[dt][v] - mu[v]) * rs[v];
      x [(size_t)t * D_ + d] = nv;
      xb[(size_t)t * D_ + d] = (__bf16)nv;
      xT[(size_t)d * T_ + t] = (__bf16)nv;
    }
  }
}

// ---------------------------------------------------------------------------
// logits = x @ lm_head  (f32 out)
__global__ void head_kernel(const __bf16* __restrict__ xb, const __bf16* __restrict__ lmT,
                            float* __restrict__ out) {
  int lane = threadIdx.x & 31, w = threadIdx.x >> 5;
  int tile = blockIdx.x * 8 + w;                  // (T/16)*(V/16) tiles
  int it = tile >> 4, nt = tile & 15;
  const __bf16* A  = xb + (size_t)(it * 16) * D_;
  const __bf16* Bt = lmT + (size_t)(nt * 16) * D_;
  v8f c = {};
#pragma unroll
  for (int kt = 0; kt < 8; ++kt)
    c = wmma_bf16(load_frag_a(A + kt * 32, D_, lane),
                  load_frag_b(Bt + kt * 32, D_, lane), c);
  int nl = lane & 15, half = lane >> 4;
#pragma unroll
  for (int v = 0; v < 8; ++v) {
    int t = it * 16 + half * 8 + v;
    out[(size_t)t * V_ + nt * 16 + nl] = c[v];
  }
}

// ---------------------------------------------------------------------------
extern "C" void kernel_launch(void* const* d_in, const int* in_sizes, int n_in,
                              void* d_out, int out_size, void* d_ws, size_t ws_size,
                              hipStream_t stream) {
  (void)in_sizes; (void)n_in; (void)out_size; (void)ws_size;
  const int*   idx   = (const int*)  d_in[0];
  const float* embed = (const float*)d_in[1];
  const float* enc   = (const float*)d_in[2];
  const float* encv  = (const float*)d_in[3];
  const float* dec   = (const float*)d_in[4];
  const float* lmh   = (const float*)d_in[5];
  const float* freqs = (const float*)d_in[6];

  char* ws = (char*)d_ws;
  size_t off = 0;
  auto alloc = [&](size_t bytes) -> void* {
    void* p = ws + off;
    off += (bytes + 255) & ~(size_t)255;
    return p;
  };
  float*  x     = (float*) alloc((size_t)T_ * D_ * 4);
  __bf16* xb    = (__bf16*)alloc((size_t)T_ * D_ * 2);
  __bf16* xT    = (__bf16*)alloc((size_t)D_ * T_ * 2);
  __bf16* encT  = (__bf16*)alloc((size_t)NH_ * N_ * D_ * 2);
  __bf16* encVT = (__bf16*)alloc((size_t)NH_ * N_ * D_ * 2);
  __bf16* decT  = (__bf16*)alloc((size_t)D_ * HN_ * 2);
  __bf16* lmT   = (__bf16*)alloc((size_t)V_ * D_ * 2);
  __bf16* xs    = (__bf16*)alloc((size_t)NH_ * T_ * N_ * 2);
  __bf16* qr    = (__bf16*)alloc((size_t)NH_ * T_ * N_ * 2);
  __bf16* ykv   = (__bf16*)alloc((size_t)NH_ * T_ * D_ * 2);
  __bf16* xy    = (__bf16*)alloc((size_t)T_ * HN_ * 2);

  // One-time weight transposes (bf16 Bt layout)
  prep_wT_kernel<<<(NH_ * D_ * N_ + 255) / 256, 256, 0, stream>>>(enc,  encT,  NH_, D_,  N_);
  prep_wT_kernel<<<(NH_ * D_ * N_ + 255) / 256, 256, 0, stream>>>(encv, encVT, NH_, D_,  N_);
  prep_wT_kernel<<<(HN_ * D_      + 255) / 256, 256, 0, stream>>>(dec,  decT,  1,   HN_, D_);
  prep_wT_kernel<<<(D_ * V_       + 255) / 256, 256, 0, stream>>>(lmh,  lmT,   1,   D_,  V_);

  embed_ln_kernel<<<T_, D_, 0, stream>>>(idx, embed, x, xb, xT);

  for (int l = 0; l < NL_; ++l) {
    enc_rope_kernel<<<NH_ * (T_ / 16) * (N_ / 16) / 8, 256, 0, stream>>>(xb, encT, freqs, xs, qr);
    attn_kernel    <<<NH_ * (T_ / 16) / QB_,          256, 0, stream>>>(qr, xT, ykv);
    encv_gate_kernel<<<NH_ * (T_ / 16) * (N_ / 16) / 8, 256, 0, stream>>>(ykv, encVT, xs, xy);
    mlp_ln_kernel  <<<(T_ / 16) / 8,                  256, 0, stream>>>(xy, decT, x, xb, xT);
  }

  head_kernel<<<(T_ / 16) * (V_ / 16) / 8, 256, 0, stream>>>(xb, lmT, (float*)d_out);
}
